// pointnet_encoder_25744033972898
// MI455X (gfx1250) — compile-verified, matrix operand order checked by reading, not run on hardware
//
#include <hip/hip_runtime.h>
#include <stdint.h>

typedef __attribute__((ext_vector_type(16))) _Float16 v16h;
typedef __attribute__((ext_vector_type(8)))  _Float16 v8h;
typedef __attribute__((ext_vector_type(8)))  float    v8f;

static inline long cdivl(long a, long b) { return (a + b - 1) / b; }
static inline int  rup(int x, int m) { return (x + m - 1) / m * m; }

// ---------------------------------------------------------------------------
// Farthest point sampling: one block per batch, dist+coords cached in regs.
// Matches jax scan semantics: emit current farthest, then update dist/argmax.
// npoint < 0 -> read from np_ptr on device (graph-capture safe).
// ---------------------------------------------------------------------------
__global__ __launch_bounds__(1024)
void fps_kernel(const float* __restrict__ xyz, int N, int npoint,
                const int* __restrict__ np_ptr, int* __restrict__ out_idx)
{
    const int b = blockIdx.x;
    const int tid = threadIdx.x;
    const int T = blockDim.x;
    const int lane = tid & 31, warp = tid >> 5;
    const int NW = T >> 5;
    __shared__ float swv[32];
    __shared__ int   swi[32];
    __shared__ int   sf;
    const int np = (npoint >= 0) ? npoint : np_ptr[0];
    const float* P = xyz + (size_t)b * N * 3;

    float px[8], py[8], pz[8], dist[8];
    int cnt = 0;
    for (int n = tid; n < N && cnt < 8; n += T) {
        px[cnt] = P[n*3+0]; py[cnt] = P[n*3+1]; pz[cnt] = P[n*3+2];
        dist[cnt] = 1e10f; cnt++;
    }
    if (tid == 0) sf = 0;
    __syncthreads();

    for (int it = 0; it < np; ++it) {
        int far = sf;
        if (tid == 0) out_idx[(size_t)b*np + it] = far;
        float cx = P[far*3+0], cy = P[far*3+1], cz = P[far*3+2];
        float bv = -1.0f; int bi = 0x7fffffff;
        #pragma unroll
        for (int t = 0; t < 8; ++t) {
            if (t < cnt) {
                float dx = px[t]-cx, dy = py[t]-cy, dz = pz[t]-cz;
                float d = dx*dx + dy*dy + dz*dz;
                float nd = fminf(dist[t], d);
                dist[t] = nd;
                int n = tid + t*T;
                if (nd > bv || (nd == bv && n < bi)) { bv = nd; bi = n; }
            }
        }
        for (int off = 16; off > 0; off >>= 1) {
            float ov = __shfl_xor(bv, off);
            int   oi = __shfl_xor(bi, off);
            if (ov > bv || (ov == bv && oi < bi)) { bv = ov; bi = oi; }
        }
        if (lane == 0) { swv[warp] = bv; swi[warp] = bi; }
        __syncthreads();
        if (warp == 0) {
            float v = (lane < NW) ? swv[lane] : -1.0f;
            int   i = (lane < NW) ? swi[lane] : 0x7fffffff;
            for (int off = 16; off > 0; off >>= 1) {
                float ov = __shfl_xor(v, off);
                int   oi = __shfl_xor(i, off);
                if (ov > v || (ov == v && oi < i)) { v = ov; i = oi; }
            }
            if (lane == 0) sf = i;
        }
        __syncthreads();
    }
}

// out[b, s, c] = src[b, idx[b*rowsPerB+s], c]   (total = B*rowsPerB*D)
__global__ void gather_kernel(const float* __restrict__ src, const int* __restrict__ idx,
                              float* __restrict__ dst, int rowsPerB, int N, int D, long total)
{
    long i = (long)blockIdx.x * blockDim.x + threadIdx.x;
    if (i >= total) return;
    int c = (int)(i % D);
    long r = i / D;
    int b = (int)(r / rowsPerB);
    int id = idx[r];
    if (id >= N) id = N - 1;
    dst[i] = src[((size_t)b*N + id)*D + c];
}

// First `nsample` in-range indices in ascending order; remainder = first hit.
__global__ void ball_query_kernel(const float* __restrict__ xyz, const float* __restrict__ new_xyz,
                                  int N, int S, float r2, int nsample,
                                  int* __restrict__ idx, int total)
{
    int i = blockIdx.x * blockDim.x + threadIdx.x;  // over B*S
    if (i >= total) return;
    int b = i / S;
    const float* P = xyz + (size_t)b * N * 3;
    float qx = new_xyz[(size_t)i*3+0], qy = new_xyz[(size_t)i*3+1], qz = new_xyz[(size_t)i*3+2];
    int* out = idx + (size_t)i * nsample;
    int found = 0;
    for (int n = 0; n < N && found < nsample; ++n) {
        float dx = P[n*3]-qx, dy = P[n*3+1]-qy, dz = P[n*3+2]-qz;
        if (dx*dx + dy*dy + dz*dz <= r2) out[found++] = n;
    }
    int fill = (found > 0) ? out[0] : 0;
    for (int k = found; k < nsample; ++k) out[k] = fill;
}

// h[row, 0:3]=xyz[idx]-new_xyz, h[row, 3:3+Cpts]=points[idx], pad->0 ; f16 out
__global__ void sa_group_kernel(const float* __restrict__ xyz, const float* __restrict__ points,
                                const float* __restrict__ new_xyz, const int* __restrict__ idx,
                                int N, int S, int nsample, int Cpts, int Kp,
                                _Float16* __restrict__ Xh, long total)
{
    long i = (long)blockIdx.x * blockDim.x + threadIdx.x;  // over B*S*ns*Kp
    if (i >= total) return;
    int c = (int)(i % Kp);
    long row = i / Kp;           // b*S*ns + s*ns + k
    long bs = row / nsample;     // b*S + s
    int b = (int)(bs / S);
    int id = idx[row];
    float v = 0.f;
    if (c < 3)
        v = xyz[((size_t)b*N + id)*3 + c] - new_xyz[bs*3 + c];
    else if (c < 3 + Cpts)
        v = points[((size_t)b*N + id)*Cpts + (c - 3)];
    Xh[i] = (_Float16)v;
}

// W (Cout,Cin) f32 -> (Np,Kp) f16, zero padded
__global__ void conv_w_kernel(const float* __restrict__ W, _Float16* __restrict__ Wh,
                              int Cout, int Cin, int Kp, long total)
{
    long i = (long)blockIdx.x * blockDim.x + threadIdx.x;
    if (i >= total) return;
    int c = (int)(i % Kp);
    int r = (int)(i / Kp);
    float v = (r < Cout && c < Cin) ? W[(size_t)r*Cin + c] : 0.f;
    Wh[i] = (_Float16)v;
}

// ---------------------------------------------------------------------------
// WMMA GEMM: Y[M,N](f32) = X[M,Kp](f16) @ W[Np,Kp](f16)^T + bias
// One wave computes a 32x64 output patch: 2 A fragments x 4 B fragments,
// each B fragment feeds 2 independent wmma ops (latency hiding + 2x B reuse).
// Per-lane layout for v_wmma_f32_16x16x32_f16 fragments:
//   A: row = lane&15, K = (lane>>4)*8 + {0..7, 16..23}  (two 16B loads)
//   B: col = lane&15, same K pattern (W rows are B columns -> contiguous K)
//   C/D vgpr v: row = v + 8*(lane>>4), col = lane&15
// Requires: M % 32 == 0 (true for every layer in this network).
// ---------------------------------------------------------------------------
__global__ __launch_bounds__(256)
void gemm_wmma_kernel(const _Float16* __restrict__ X, const _Float16* __restrict__ W,
                      const float* __restrict__ bias, float* __restrict__ Y,
                      int M, int N, int Kp, int tilesM32, int tilesN64)
{
    int wave = blockIdx.x * (blockDim.x >> 5) + (threadIdx.x >> 5);
    if (wave >= tilesM32 * tilesN64) return;   // wave-uniform: EXEC stays all-1
    int lane = threadIdx.x & 31;
    int tn = wave % tilesN64;
    int tm = wave / tilesN64;
    int hsel = lane >> 4;
    int l15  = lane & 15;

    const _Float16* xrow0 = X + (size_t)(tm*32 + l15)      * Kp + hsel*8;
    const _Float16* xrow1 = X + (size_t)(tm*32 + 16 + l15) * Kp + hsel*8;
    const _Float16* wbase = W + (size_t)(tn*64 + l15)      * Kp + hsel*8;

    v8f acc[2][4] = {};
    for (int k0 = 0; k0 < Kp; k0 += 32) {
        if (k0 + 32 < Kp) {
            __builtin_prefetch(xrow0 + k0 + 32, 0, 3);
            __builtin_prefetch(xrow1 + k0 + 32, 0, 3);
            __builtin_prefetch(wbase + k0 + 32, 0, 3);
        }
        v8h a0lo = *(const v8h*)(xrow0 + k0);
        v8h a0hi = *(const v8h*)(xrow0 + k0 + 16);
        v8h a1lo = *(const v8h*)(xrow1 + k0);
        v8h a1hi = *(const v8h*)(xrow1 + k0 + 16);
        v16h a0, a1;
        #pragma unroll
        for (int t = 0; t < 8; ++t) {
            a0[t] = a0lo[t]; a0[t+8] = a0hi[t];
            a1[t] = a1lo[t]; a1[t+8] = a1hi[t];
        }
        #pragma unroll
        for (int j = 0; j < 4; ++j) {
            const _Float16* wp = wbase + (size_t)j*16*Kp + k0;
            v8h blo = *(const v8h*)(wp);
            v8h bhi = *(const v8h*)(wp + 16);
            v16h bb;
            #pragma unroll
            for (int t = 0; t < 8; ++t) { bb[t] = blo[t]; bb[t+8] = bhi[t]; }
            acc[0][j] = __builtin_amdgcn_wmma_f32_16x16x32_f16(
                false, a0, false, bb, (short)0, acc[0][j], false, false);
            acc[1][j] = __builtin_amdgcn_wmma_f32_16x16x32_f16(
                false, a1, false, bb, (short)0, acc[1][j], false, false);
        }
    }
    #pragma unroll
    for (int i = 0; i < 2; ++i) {
        int row0 = tm*32 + i*16 + hsel*8;
        #pragma unroll
        for (int j = 0; j < 4; ++j) {
            int col = tn*64 + j*16 + l15;
            if (col < N) {
                float bv = bias ? bias[col] : 0.f;
                #pragma unroll
                for (int v = 0; v < 8; ++v)
                    Y[(size_t)(row0 + v)*N + col] = acc[i][j][v] + bv;
            }
        }
    }
}

__global__ void fill_f32_kernel(float* __restrict__ p, float val, long n)
{
    long i = (long)blockIdx.x * blockDim.x + threadIdx.x;
    if (i < n) p[i] = val;
}

// per-channel sum / sumsq -> stats[c], stats[N+c]
__global__ __launch_bounds__(256)
void bn_stats_kernel(const float* __restrict__ Y, int M, int N, int rowsPerBlk,
                     float* __restrict__ stats)
{
    int c = blockIdx.x;
    int m0 = blockIdx.y * rowsPerBlk;
    int m1 = m0 + rowsPerBlk; if (m1 > M) m1 = M;
    float s = 0.f, s2 = 0.f;
    for (int m = m0 + threadIdx.x; m < m1; m += blockDim.x) {
        float v = Y[(size_t)m*N + c];
        s += v; s2 += v*v;
    }
    for (int off = 16; off > 0; off >>= 1) { s += __shfl_xor(s, off); s2 += __shfl_xor(s2, off); }
    __shared__ float sh[2][8];
    int warp = threadIdx.x >> 5, lane = threadIdx.x & 31;
    if (lane == 0) { sh[0][warp] = s; sh[1][warp] = s2; }
    __syncthreads();
    if (threadIdx.x == 0) {
        float t = 0.f, t2 = 0.f;
        for (int w = 0; w < (int)(blockDim.x >> 5); ++w) { t += sh[0][w]; t2 += sh[1][w]; }
        atomicAdd(&stats[c], t);
        atomicAdd(&stats[N + c], t2);
    }
}

// stats -> (scale, shift) in place
__global__ void bn_finalize_kernel(float* __restrict__ stats, const float* __restrict__ gamma,
                                   const float* __restrict__ beta, int N, float invM)
{
    int c = blockIdx.x * blockDim.x + threadIdx.x;
    if (c >= N) return;
    float mean = stats[c] * invM;
    float var  = stats[N + c] * invM - mean * mean;
    float sc = gamma[c] * rsqrtf(var + 1e-5f);
    stats[c]     = sc;
    stats[N + c] = beta[c] - mean * sc;
}

// relu(y*scale+shift) -> optional f32 (stride N) and/or padded f16 (stride Kp)
__global__ void bn_apply_kernel(const float* __restrict__ Y, const float* __restrict__ stats,
                                int M, int N, int Kp,
                                float* __restrict__ outF, _Float16* __restrict__ outH, long total)
{
    long i = (long)blockIdx.x * blockDim.x + threadIdx.x;  // over M*Kp
    if (i >= total) return;
    int c = (int)(i % Kp);
    long m = i / Kp;
    if (c < N) {
        float v = Y[(size_t)m*N + c];
        v = fmaxf(v * stats[c] + stats[N + c], 0.f);
        if (outF) outF[(size_t)m*N + c] = v;
        if (outH) outH[i] = (_Float16)v;
    } else if (outH) {
        outH[i] = (_Float16)0.f;
    }
}

// out[bs, c] = max_k Y[bs*nsample + k, c]
__global__ void maxpool_kernel(const float* __restrict__ Y, int nsample, int C,
                               float* __restrict__ out, long total)
{
    long i = (long)blockIdx.x * blockDim.x + threadIdx.x;
    if (i >= total) return;
    int c = (int)(i % C);
    long s = i / C;
    const float* p = Y + (size_t)s * nsample * C + c;
    float m = p[0];
    for (int k = 1; k < nsample; ++k) m = fmaxf(m, p[(size_t)k * C]);
    out[i] = m;
}

// 3-NN (ascending distance, lower index wins ties) + normalized inverse-dist w
__global__ void knn3_kernel(const float* __restrict__ xyz1, const float* __restrict__ xyz2,
                            int N1, int S2, int* __restrict__ idx3, float* __restrict__ w3,
                            int total)
{
    int i = blockIdx.x * blockDim.x + threadIdx.x;  // over B*N1
    if (i >= total) return;
    int b = i / N1;
    float qx = xyz1[(size_t)i*3], qy = xyz1[(size_t)i*3+1], qz = xyz1[(size_t)i*3+2];
    const float* P = xyz2 + (size_t)b * S2 * 3;
    float d0 = 3.4e38f, d1 = 3.4e38f, d2 = 3.4e38f;
    int   i0 = 0, i1 = 0, i2 = 0;
    for (int s = 0; s < S2; ++s) {
        float dx = P[s*3]-qx, dy = P[s*3+1]-qy, dz = P[s*3+2]-qz;
        float d = dx*dx + dy*dy + dz*dz;
        if (d < d0)      { d2=d1; i2=i1; d1=d0; i1=i0; d0=d; i0=s; }
        else if (d < d1) { d2=d1; i2=i1; d1=d;  i1=s; }
        else if (d < d2) { d2=d;  i2=s; }
    }
    float w0 = 1.f/(d0+1e-8f), w1 = 1.f/(d1+1e-8f), w2 = 1.f/(d2+1e-8f);
    float ws = w0 + w1 + w2;
    w3[(size_t)i*3+0] = w0/ws; w3[(size_t)i*3+1] = w1/ws; w3[(size_t)i*3+2] = w2/ws;
    idx3[(size_t)i*3+0] = i0; idx3[(size_t)i*3+1] = i1; idx3[(size_t)i*3+2] = i2;
}

// h[row, 0:C1]=pts1 ; h[row, C1:C1+C2]=sum_j w_j*pts2[idx_j] ; pad->0 ; f16 out
__global__ void fp_concat_kernel(const float* __restrict__ pts1, const float* __restrict__ pts2,
                                 const int* __restrict__ idx3, const float* __restrict__ w3,
                                 int N1, int S2, int C1, int C2, int Kp,
                                 _Float16* __restrict__ Xh, long total)
{
    long i = (long)blockIdx.x * blockDim.x + threadIdx.x;  // over B*N1*Kp
    if (i >= total) return;
    int c = (int)(i % Kp);
    long r = i / Kp;              // b*N1 + n
    int b = (int)(r / N1);
    float v = 0.f;
    if (c < C1) {
        v = pts1[(size_t)r*C1 + c];
    } else if (c < C1 + C2) {
        int cc = c - C1;
        const float* P2 = pts2 + (size_t)b * S2 * C2;
        v = w3[(size_t)r*3+0] * P2[(size_t)idx3[(size_t)r*3+0]*C2 + cc]
          + w3[(size_t)r*3+1] * P2[(size_t)idx3[(size_t)r*3+1]*C2 + cc]
          + w3[(size_t)r*3+2] * P2[(size_t)idx3[(size_t)r*3+2]*C2 + cc];
    }
    Xh[i] = (_Float16)v;
}

// ===========================================================================
extern "C" void kernel_launch(void* const* d_in, const int* in_sizes, int n_in,
                              void* d_out, int out_size, void* d_ws, size_t ws_size,
                              hipStream_t stream)
{
    (void)in_sizes; (void)n_in; (void)ws_size;
    const int B = 2;
    const float* xyz0 = (const float*)d_in[0];
    const int* num_samples_dev = (const int*)d_in[1];

    // ---- workspace arena ----
    char* ws = (char*)d_ws;
    size_t off = 0;
    auto alloc = [&](size_t bytes) -> void* {
        off = (off + 255) & ~(size_t)255;
        void* p = ws + off; off += bytes; return p;
    };
    float* l1_xyz = (float*)alloc((size_t)B*4096*3*4);
    float* l2_xyz = (float*)alloc((size_t)B*2048*3*4);
    float* l3_xyz = (float*)alloc((size_t)B*1024*3*4);
    float* l4_xyz = (float*)alloc((size_t)B*256*3*4);
    float* l1_pts = (float*)alloc((size_t)B*4096*64*4);
    float* l2_pts = (float*)alloc((size_t)B*2048*128*4);
    float* l3_pts = (float*)alloc((size_t)B*1024*256*4);
    float* l4_pts = (float*)alloc((size_t)B*256*512*4);
    float* fp4_out = (float*)alloc((size_t)B*1024*512*4);
    float* fp3_out = (float*)alloc((size_t)B*2048*1024*4);
    float* fp2_out = (float*)alloc((size_t)B*4096*1024*4);
    float* xhead   = (float*)alloc((size_t)B*8192*3*4);
    int*   fpsidx  = (int*)alloc((size_t)B*4096*4);
    int*   ballidx = (int*)alloc((size_t)B*4096*32*4);
    int*   knn_idx = (int*)alloc((size_t)B*8192*3*4);
    float* knn_w   = (float*)alloc((size_t)B*8192*3*4);
    float* stats   = (float*)alloc(2048*4);
    float* Ybuf    = (float*)alloc((size_t)16777216*4);
    _Float16* xh_a = (_Float16*)alloc((size_t)16777216*2);
    _Float16* xh_b = (_Float16*)alloc((size_t)16777216*2);
    _Float16* Wh   = (_Float16*)alloc((size_t)1114112*2);

    auto g1 = [](long tot) { return dim3((unsigned)cdivl(tot, 256)); };

    // ---- MLP chain: gemm + BN(+relu) per layer ----
    // pbase: d_in index of first (w,b,g,be); returns f16 buffer of final output
    auto run_mlp = [&](int M, const int* dims, int nl, int pbase,
                       _Float16* xin, float* finalF32, bool wantFinalHalf) -> _Float16* {
        _Float16* cur = xin;
        _Float16* lastH = nullptr;
        for (int l = 0; l < nl; ++l) {
            int Cin = dims[l], Cout = dims[l+1];
            int Kp = rup(Cin, 32), Np = rup(Cout, 64);
            const float* w  = (const float*)d_in[pbase + l*4 + 0];
            const float* bi = (const float*)d_in[pbase + l*4 + 1];
            const float* ga = (const float*)d_in[pbase + l*4 + 2];
            const float* be = (const float*)d_in[pbase + l*4 + 3];
            long wtot = (long)Np * Kp;
            conv_w_kernel<<<g1(wtot), 256, 0, stream>>>(w, Wh, Cout, Cin, Kp, wtot);
            int tilesM32 = M / 32, tilesN64 = Np / 64;
            long waves = (long)tilesM32 * tilesN64;
            gemm_wmma_kernel<<<(unsigned)cdivl(waves, 8), 256, 0, stream>>>(
                cur, Wh, bi, Ybuf, M, Cout, Kp, tilesM32, tilesN64);
            fill_f32_kernel<<<g1(2*Cout), 256, 0, stream>>>(stats, 0.f, 2*Cout);
            int rowsPerBlk = 8192;
            int chunks = (int)cdivl(M, rowsPerBlk);
            bn_stats_kernel<<<dim3(Cout, chunks), 256, 0, stream>>>(Ybuf, M, Cout, rowsPerBlk, stats);
            bn_finalize_kernel<<<g1(Cout), 256, 0, stream>>>(stats, ga, be, Cout, 1.f / (float)M);
            bool last = (l == nl - 1);
            _Float16* nxt = (cur == xh_a) ? xh_b : xh_a;
            int KpOut = rup(Cout, 32);
            float*    oF = last ? finalF32 : nullptr;
            _Float16* oH = last ? (wantFinalHalf ? nxt : nullptr) : nxt;
            if (oF || oH) {
                long tot = (long)M * KpOut;
                bn_apply_kernel<<<g1(tot), 256, 0, stream>>>(Ybuf, stats, M, Cout, KpOut, oF, oH, tot);
            }
            if (last) lastH = oH;
            cur = nxt;
        }
        return lastH;
    };

    // ---- set abstraction ----
    auto run_sa = [&](const float* in_xyz, const float* in_pts, int Nin, int Cpts,
                      int Snp, float radius, int pbase, const int* dims, int nl,
                      float* out_xyz, float* out_pts, int Cfinal) {
        fps_kernel<<<B, 1024, 0, stream>>>(in_xyz, Nin, Snp, nullptr, fpsidx);
        long gt = (long)B * Snp * 3;
        gather_kernel<<<g1(gt), 256, 0, stream>>>(in_xyz, fpsidx, out_xyz, Snp, Nin, 3, gt);
        ball_query_kernel<<<g1((long)B*Snp), 256, 0, stream>>>(
            in_xyz, out_xyz, Nin, Snp, radius*radius, 32, ballidx, B*Snp);
        int Kp0 = rup(3 + Cpts, 32);
        long tg = (long)B * Snp * 32 * Kp0;
        sa_group_kernel<<<g1(tg), 256, 0, stream>>>(
            in_xyz, in_pts, out_xyz, ballidx, Nin, Snp, 32, Cpts, Kp0, xh_a, tg);
        int M = B * Snp * 32;
        run_mlp(M, dims, nl, pbase, xh_a, /*finalF32 in-place=*/Ybuf, false);
        long tp = (long)B * Snp * Cfinal;
        maxpool_kernel<<<g1(tp), 256, 0, stream>>>(Ybuf, 32, Cfinal, out_pts, tp);
    };

    // ---- feature propagation ----
    auto run_fp = [&](const float* xyz1, const float* xyz2, int N1, int S2,
                      const float* pts1, int C1, const float* pts2, int C2,
                      int pbase, const int* dims, int nl,
                      float* finalF32, bool wantHalf) -> _Float16* {
        knn3_kernel<<<g1((long)B*N1), 256, 0, stream>>>(xyz1, xyz2, N1, S2, knn_idx, knn_w, B*N1);
        int Kp = rup(C1 + C2, 32);
        long tot = (long)B * N1 * Kp;
        fp_concat_kernel<<<g1(tot), 256, 0, stream>>>(
            pts1, pts2, knn_idx, knn_w, N1, S2, C1, C2, Kp, xh_a, tot);
        return run_mlp(B * N1, dims, nl, pbase, xh_a, finalF32, wantHalf);
    };

    // ===================== pipeline =====================
    const int d_sa1[4] = {6, 32, 32, 64};
    const int d_sa2[4] = {67, 64, 64, 128};
    const int d_sa3[4] = {131, 128, 128, 256};
    const int d_sa4[4] = {259, 256, 256, 512};
    run_sa(xyz0,   xyz0,   8192, 3,   4096, 0.2f, 2,  d_sa1, 3, l1_xyz, l1_pts, 64);
    run_sa(l1_xyz, l1_pts, 4096, 64,  2048, 0.4f, 14, d_sa2, 3, l2_xyz, l2_pts, 128);
    run_sa(l2_xyz, l2_pts, 2048, 128, 1024, 0.8f, 26, d_sa3, 3, l3_xyz, l3_pts, 256);
    run_sa(l3_xyz, l3_pts, 1024, 256, 256,  1.0f, 38, d_sa4, 3, l4_xyz, l4_pts, 512);

    const int d_fp4[3] = {768, 256, 512};
    const int d_fp3[3] = {640, 512, 1024};
    const int d_fp2[3] = {1088, 1024, 1024};
    const int d_fp1[4] = {1024, 1024, 1024, 1024};
    run_fp(l3_xyz, l4_xyz, 1024, 256,  l3_pts, 256, l4_pts,  512,  50, d_fp4, 2, fp4_out, false);
    run_fp(l2_xyz, l3_xyz, 2048, 1024, l2_pts, 128, fp4_out, 512,  58, d_fp3, 2, fp3_out, false);
    run_fp(l1_xyz, l2_xyz, 4096, 2048, l1_pts, 64,  fp3_out, 1024, 66, d_fp2, 2, fp2_out, false);
    _Float16* h0 = run_fp(xyz0, l1_xyz, 8192, 4096, nullptr, 0, fp2_out, 1024,
                          74, d_fp1, 3, nullptr, true);

    // ---- head: two BN layers + final linear (no BN) ----
    const int d_head[3] = {1024, 1024, 512};
    _Float16* h2 = run_mlp(B * 8192, d_head, 2, 86, h0, nullptr, true);
    {
        int Cin = 512, Cout = 3;
        int Kp = rup(Cin, 32), Np = rup(Cout, 64);
        long wtot = (long)Np * Kp;
        conv_w_kernel<<<g1(wtot), 256, 0, stream>>>((const float*)d_in[94], Wh, Cout, Cin, Kp, wtot);
        int M = B * 8192, tilesM32 = M / 32, tilesN64 = Np / 64;
        long waves = (long)tilesM32 * tilesN64;
        gemm_wmma_kernel<<<(unsigned)cdivl(waves, 8), 256, 0, stream>>>(
            h2, Wh, (const float*)d_in[95], xhead, M, Cout, Kp, tilesM32, tilesN64);
    }

    // ---- final FPS on x (npoint read on-device) + gather into d_out ----
    fps_kernel<<<B, 1024, 0, stream>>>(xhead, 8192, -1, num_samples_dev, fpsidx);
    int rowsPerB = (out_size / 3) / B;
    gather_kernel<<<g1((long)out_size), 256, 0, stream>>>(
        xhead, fpsidx, (float*)d_out, rowsPerB, 8192, 3, out_size);
}
